// GPT_62251255989204
// MI455X (gfx1250) — compile-verified, matrix operand order checked by reading, not run on hardware
//

#include <hip/hip_runtime.h>
#include <hip/hip_bf16.h>
#include <math.h>
#include <stdint.h>

// ---------------------------------------------------------------------------
// GPT-2 small forward for gfx1250 (MI455X). All matmuls via
// v_wmma_f32_16x16x32_f16 (wave32, f32 accumulate). Each GEMM wave owns a
// 16(M) x 64(N) C footprint: one A fragment feeds 4 WMMAs per K-step.
// A tile staged to LDS with global_load_async_to_lds_b128 (ASYNCcnt path),
// next K-panel of W prefetched via global_prefetch_b8.
// Model dims: L=12 H=12 D=768 DH=64 V=50257 B=4 T=512 (TV=64 + TT=448).
// ---------------------------------------------------------------------------

#define Lcfg 12
#define Hcfg 12
#define Dcfg 768
#define DH   64
#define Vcfg 50257
#define Bcfg 4
#define TV   64
#define TT   448
#define Tcfg 512
#define Mrows (Bcfg * Tcfg)   // 2048

typedef __attribute__((ext_vector_type(16))) _Float16 v16h;
typedef __attribute__((ext_vector_type(8)))  float    v8f;

// CDNA5 16-bit A/B fragment K index for VGPR pair v (0..7), lane-half kh (0/1):
// VGPR0..3 hold K = kh*8 + {0,2,4,6}(+1), VGPR4..7 hold K = 16 + kh*8 + {0,2,4,6}(+1).
__device__ __forceinline__ int frag_k(int v, int kh) {
    return ((v < 4) ? (2 * v) : (16 + 2 * (v - 4))) + kh * 8;
}

__device__ __forceinline__ v8f wmma16(v16h a, v16h b, v8f c) {
    return __builtin_amdgcn_wmma_f32_16x16x32_f16(
        false, a, false, b, (short)0, c, false, false);
}

// ---------------------------------------------------------------------------
// Embedding: h[b,t,:] = (t<TV ? visual[b,t,:] : wte[x[b,t-TV],:]) + wpe[t,:]
// ---------------------------------------------------------------------------
__global__ void embed_kernel(const int* __restrict__ x,
                             const float* __restrict__ vis,
                             const float* __restrict__ wte,
                             const float* __restrict__ wpe,
                             float* __restrict__ h) {
    const int t = blockIdx.x;
    const int b = blockIdx.y;
    for (int d = threadIdx.x; d < Dcfg; d += blockDim.x) {
        float e;
        if (t < TV) {
            e = vis[((size_t)b * TV + t) * Dcfg + d];
        } else {
            int tok = x[b * TT + (t - TV)];
            e = wte[(size_t)tok * Dcfg + d];
        }
        h[((size_t)b * Tcfg + t) * Dcfg + d] = e + wpe[(size_t)t * Dcfg + d];
    }
}

// ---------------------------------------------------------------------------
// LayerNorm over last dim (D=768). One 256-thread block per row.
// ---------------------------------------------------------------------------
__global__ void ln_kernel(const float* __restrict__ in,
                          const float* __restrict__ w,
                          const float* __restrict__ b,
                          float* __restrict__ out) {
    __shared__ float red[256];
    const int row = blockIdx.x;
    const int tid = threadIdx.x;
    const float* xr = in + (size_t)row * Dcfg;

    float s = 0.f;
    for (int d = tid; d < Dcfg; d += 256) s += xr[d];
    red[tid] = s; __syncthreads();
    for (int o = 128; o > 0; o >>= 1) {
        if (tid < o) red[tid] += red[tid + o];
        __syncthreads();
    }
    const float mu = red[0] * (1.0f / Dcfg);
    __syncthreads();

    float v = 0.f;
    for (int d = tid; d < Dcfg; d += 256) { float t = xr[d] - mu; v += t * t; }
    red[tid] = v; __syncthreads();
    for (int o = 128; o > 0; o >>= 1) {
        if (tid < o) red[tid] += red[tid + o];
        __syncthreads();
    }
    const float rstd = rsqrtf(red[0] * (1.0f / Dcfg) + 1e-5f);

    for (int d = tid; d < Dcfg; d += 256)
        out[(size_t)row * Dcfg + d] = (xr[d] - mu) * rstd * w[d] + b[d];
}

// ---------------------------------------------------------------------------
// Generic WMMA GEMM: C[M,N] = act(A[M,K] @ W + bias) (+ residual)
//   BT=0: W is [K,N] row-major (ldb=N).  BT=1: W is [N,K] row-major (ldb=K).
//   ACT=1: exact GELU. RES=1: add residual.
// Block = 128 threads = 4 waves; block tile 16(M) x 256(N);
// wave tile 16 x 64 (4 accumulators), one shared A fragment per K-step.
// A tile (16x32 f32) staged via async global->LDS DMA, shared by all 4 waves.
// Row pad = 36 floats: 16B-aligned rows (144B) AND 16 distinct LDS banks for
// the column-broadcast fragment reads (36*r mod 64 distinct for r=0..15).
// ---------------------------------------------------------------------------
template <int BT, int ACT, int RES>
__global__ void gemm_wmma_kernel(const float* __restrict__ A, int lda,
                                 const float* __restrict__ W, int ldb,
                                 const float* __restrict__ bias,
                                 const float* __restrict__ res, int ldr,
                                 float* __restrict__ C, int ldc,
                                 int N, int K) {
    __shared__ float As[16][36];

    const int tid  = threadIdx.x;
    const int wave = tid >> 5;
    const int lane = tid & 31;
    const int m0   = blockIdx.y * 16;
    const int nb   = blockIdx.x * 256 + wave * 64;   // wave's N base (64 wide)
    const int ml   = lane & 15;
    const int kh   = lane >> 4;

    // Per-thread async-copy chunk: 16B (float4). 128 chunks == 128 threads.
    const int ar  = tid >> 3;         // A-tile row 0..15
    const int ac4 = (tid & 7) * 4;    // A-tile col 0,4,...,28
    const unsigned ldsa = (unsigned)(uintptr_t)(&As[ar][ac4]);

    v8f acc[4] = {};
    for (int k0 = 0; k0 < K; k0 += 32) {
        __syncthreads();   // previous iteration's readers are done

        // Async DMA: global A[m0+ar][k0+ac4 .. +3] -> LDS (no VGPR round trip)
        {
            const unsigned long long ga = (unsigned long long)(uintptr_t)
                (A + (size_t)(m0 + ar) * lda + (k0 + ac4));
            asm volatile("global_load_async_to_lds_b128 %0, %1, off"
                         :: "v"(ldsa), "v"(ga) : "memory");
            asm volatile("s_wait_asynccnt 0x0" ::: "memory");
        }
        __syncthreads();   // tile published to all 4 waves

        // Prefetch next K-panel of W into cache (global_prefetch_b8).
        if (k0 + 32 < K) {
            if (BT) {
                const int pr = blockIdx.x * 256 + 2 * tid;   // 2 rows per thread
                if (pr < N)     __builtin_prefetch(W + (size_t)pr * ldb + (k0 + 32), 0, 1);
                if (pr + 1 < N) __builtin_prefetch(W + (size_t)(pr + 1) * ldb + (k0 + 32), 0, 1);
            } else {
                const int pr = k0 + 32 + (tid >> 2);          // 32 rows
                const int pc = blockIdx.x * 256 + (tid & 3) * 64;
                if (pc < N) __builtin_prefetch(W + (size_t)pr * ldb + pc, 0, 1);
            }
        }

        // A fragment: built once, feeds 4 WMMAs.
        v16h af;
#pragma unroll
        for (int v = 0; v < 8; ++v) {
            const int kb = frag_k(v, kh);
            af[2 * v]     = (_Float16)As[ml][kb];
            af[2 * v + 1] = (_Float16)As[ml][kb + 1];
        }

#pragma unroll
        for (int t = 0; t < 4; ++t) {
            const int nn = nb + t * 16 + ml;
            v16h bf;
#pragma unroll
            for (int v = 0; v < 8; ++v) {
                const int kb = frag_k(v, kh);
                float b0 = 0.f, b1 = 0.f;
                if (nn < N) {
                    if (BT) {
                        const float2 p = *(const float2*)(W + (size_t)nn * ldb + (k0 + kb));
                        b0 = p.x; b1 = p.y;
                    } else {
                        b0 = W[(size_t)(k0 + kb) * ldb + nn];
                        b1 = W[(size_t)(k0 + kb + 1) * ldb + nn];
                    }
                }
                bf[2 * v]     = (_Float16)b0;
                bf[2 * v + 1] = (_Float16)b1;
            }
            acc[t] = wmma16(af, bf, acc[t]);
        }
    }

#pragma unroll
    for (int t = 0; t < 4; ++t) {
        const int nn = nb + t * 16 + ml;
        if (nn < N) {
            const float bv = bias ? bias[nn] : 0.0f;
#pragma unroll
            for (int r = 0; r < 8; ++r) {
                const int mm = m0 + r + 8 * kh;
                float xv = acc[t][r] + bv;
                if (ACT == 1) xv = 0.5f * xv * (1.0f + erff(xv * 0.70710678118654752f));
                if (RES) xv += res[(size_t)mm * ldr + nn];
                C[(size_t)mm * ldc + nn] = xv;
            }
        }
    }
}

// ---------------------------------------------------------------------------
// Attention scores: S[bh,q,k] = (Q·K)/8 + (k<=q ? 1 : 0)  (additive mask!)
// grid: x = T/64 key blocks, y = T/16 q tiles, z = B*H. Block 128 = 4 waves.
// ---------------------------------------------------------------------------
__global__ void attn_scores_kernel(const float* __restrict__ qkv,
                                   float* __restrict__ att) {
    const int bh = blockIdx.z;
    const int b  = bh / Hcfg;
    const int h  = bh % Hcfg;
    const int tid = threadIdx.x, wave = tid >> 5, lane = tid & 31;
    const int q0 = blockIdx.y * 16;
    const int n0 = blockIdx.x * 64 + wave * 16;
    const int ml = lane & 15, kh = lane >> 4;
    const size_t rs = 3 * Dcfg;

    const float* Q  = qkv + (size_t)b * Tcfg * rs + (size_t)h * DH;
    const float* Kp = qkv + (size_t)b * Tcfg * rs + Dcfg + (size_t)h * DH;

    v8f acc = {};
#pragma unroll
    for (int k0 = 0; k0 < DH; k0 += 32) {
        v16h af, bf;
#pragma unroll
        for (int v = 0; v < 8; ++v) {
            const int kb = frag_k(v, kh) + k0;
            const float2 qa = *(const float2*)(Q  + (size_t)(q0 + ml) * rs + kb);
            const float2 ka = *(const float2*)(Kp + (size_t)(n0 + ml) * rs + kb);
            af[2 * v] = (_Float16)qa.x; af[2 * v + 1] = (_Float16)qa.y;
            bf[2 * v] = (_Float16)ka.x; bf[2 * v + 1] = (_Float16)ka.y;
        }
        acc = wmma16(af, bf, acc);
    }

    const float scale = 0.125f;   // 1/sqrt(64)
#pragma unroll
    for (int r = 0; r < 8; ++r) {
        const int q = q0 + r + 8 * kh;
        const int k = n0 + ml;
        att[((size_t)bh * Tcfg + q) * Tcfg + k] =
            acc[r] * scale + ((k <= q) ? 1.0f : 0.0f);
    }
}

// ---------------------------------------------------------------------------
// Stable softmax over T=512 keys; one 256-thread block per (bh,q) row.
// ---------------------------------------------------------------------------
__global__ void softmax_kernel(float* __restrict__ att) {
    __shared__ float red[256];
    const int tid = threadIdx.x;
    float* p = att + (size_t)blockIdx.x * Tcfg;

    float mx = -1e30f;
    for (int k = tid; k < Tcfg; k += 256) mx = fmaxf(mx, p[k]);
    red[tid] = mx; __syncthreads();
    for (int o = 128; o > 0; o >>= 1) {
        if (tid < o) red[tid] = fmaxf(red[tid], red[tid + o]);
        __syncthreads();
    }
    mx = red[0]; __syncthreads();

    float s = 0.f;
    for (int k = tid; k < Tcfg; k += 256) {
        float e = expf(p[k] - mx);
        p[k] = e; s += e;
    }
    red[tid] = s; __syncthreads();
    for (int o = 128; o > 0; o >>= 1) {
        if (tid < o) red[tid] += red[tid + o];
        __syncthreads();
    }
    const float inv = 1.0f / red[0];
    for (int k = tid; k < Tcfg; k += 256) p[k] *= inv;
}

// ---------------------------------------------------------------------------
// O = P @ V, merged heads: o[b,q,h*64+d]. grid: y = T/16 q tiles, z = B*H.
// Block 128 = 4 waves covering all 64 head dims. K loop over 512 keys.
// ---------------------------------------------------------------------------
__global__ void attn_av_kernel(const float* __restrict__ att,
                               const float* __restrict__ qkv,
                               float* __restrict__ o) {
    const int bh = blockIdx.z;
    const int b  = bh / Hcfg;
    const int h  = bh % Hcfg;
    const int tid = threadIdx.x, wave = tid >> 5, lane = tid & 31;
    const int q0 = blockIdx.y * 16;
    const int n0 = wave * 16;
    const int ml = lane & 15, kh = lane >> 4;
    const size_t rs = 3 * Dcfg;

    const float* P = att + (size_t)bh * Tcfg * Tcfg;
    const float* V = qkv + (size_t)b * Tcfg * rs + 2 * Dcfg + (size_t)h * DH;

    v8f acc = {};
    for (int k0 = 0; k0 < Tcfg; k0 += 32) {
        v16h af, bf;
#pragma unroll
        for (int v = 0; v < 8; ++v) {
            const int kb = frag_k(v, kh) + k0;
            const float2 pa = *(const float2*)(P + (size_t)(q0 + ml) * Tcfg + kb);
            af[2 * v] = (_Float16)pa.x; af[2 * v + 1] = (_Float16)pa.y;
            bf[2 * v]     = (_Float16)V[(size_t)kb * rs + (n0 + ml)];
            bf[2 * v + 1] = (_Float16)V[(size_t)(kb + 1) * rs + (n0 + ml)];
        }
        acc = wmma16(af, bf, acc);
    }

#pragma unroll
    for (int r = 0; r < 8; ++r) {
        const int q = q0 + r + 8 * kh;
        o[((size_t)b * Tcfg + q) * Dcfg + h * DH + (n0 + ml)] = acc[r];
    }
}

// ---------------------------------------------------------------------------
// Host-side orchestration
// ---------------------------------------------------------------------------
extern "C" void kernel_launch(void* const* d_in, const int* in_sizes, int n_in,
                              void* d_out, int out_size, void* d_ws, size_t ws_size,
                              hipStream_t stream) {
    (void)in_sizes; (void)n_in; (void)out_size; (void)ws_size;

    const int*   x      = (const int*)  d_in[0];
    const float* vis    = (const float*)d_in[1];
    const float* wte    = (const float*)d_in[2];
    const float* wpe    = (const float*)d_in[3];
    const float* ln1_w  = (const float*)d_in[4];
    const float* ln1_b  = (const float*)d_in[5];
    const float* attn_w = (const float*)d_in[6];
    const float* attn_b = (const float*)d_in[7];
    const float* proj_w = (const float*)d_in[8];
    const float* proj_b = (const float*)d_in[9];
    const float* ln2_w  = (const float*)d_in[10];
    const float* ln2_b  = (const float*)d_in[11];
    const float* fc_w   = (const float*)d_in[12];
    const float* fc_b   = (const float*)d_in[13];
    const float* fc2_w  = (const float*)d_in[14];
    const float* fc2_b  = (const float*)d_in[15];
    const float* lnf_w  = (const float*)d_in[16];
    const float* lnf_b  = (const float*)d_in[17];
    float* logits = (float*)d_out;

    // Workspace bump allocation (f32):
    float* ws   = (float*)d_ws;
    float* h    = ws;                              // [2048, 768]
    float* nbuf = h    + (size_t)Mrows * Dcfg;     // [2048, 768]
    float* qkvb = nbuf + (size_t)Mrows * Dcfg;     // [2048, 2304]
    float* att  = qkvb + (size_t)Mrows * 3 * Dcfg; // [48, 512, 512]
    float* ob   = att  + (size_t)Bcfg * Hcfg * Tcfg * Tcfg; // [2048, 768]
    float* mb   = ob   + (size_t)Mrows * Dcfg;     // [2048, 3072]

    const dim3 blk128(128);

    // Embeddings
    embed_kernel<<<dim3(Tcfg, Bcfg), 256, 0, stream>>>(x, vis, wte, wpe, h);

    for (int l = 0; l < Lcfg; ++l) {
        const float* aw  = attn_w + (size_t)l * Dcfg * 3 * Dcfg;
        const float* ab  = attn_b + (size_t)l * 3 * Dcfg;
        const float* pw  = proj_w + (size_t)l * Dcfg * Dcfg;
        const float* pb  = proj_b + (size_t)l * Dcfg;
        const float* fw  = fc_w   + (size_t)l * Dcfg * 4 * Dcfg;
        const float* fb  = fc_b   + (size_t)l * 4 * Dcfg;
        const float* f2w = fc2_w  + (size_t)l * 4 * Dcfg * Dcfg;
        const float* f2b = fc2_b  + (size_t)l * Dcfg;

        // n = LN1(h)
        ln_kernel<<<Mrows, 256, 0, stream>>>(h, ln1_w + l * Dcfg, ln1_b + l * Dcfg, nbuf);

        // qkv = n @ aw + ab        [2048,768]x[768,2304]
        gemm_wmma_kernel<0, 0, 0><<<dim3(3 * Dcfg / 256, Mrows / 16), blk128, 0, stream>>>(
            nbuf, Dcfg, aw, 3 * Dcfg, ab, nullptr, 0, qkvb, 3 * Dcfg, 3 * Dcfg, Dcfg);

        // attention
        attn_scores_kernel<<<dim3(Tcfg / 64, Tcfg / 16, Bcfg * Hcfg), blk128, 0, stream>>>(qkvb, att);
        softmax_kernel<<<Bcfg * Hcfg * Tcfg, 256, 0, stream>>>(att);
        attn_av_kernel<<<dim3(1, Tcfg / 16, Bcfg * Hcfg), blk128, 0, stream>>>(att, qkvb, ob);

        // h = h + o @ pw + pb
        gemm_wmma_kernel<0, 0, 1><<<dim3(Dcfg / 256, Mrows / 16), blk128, 0, stream>>>(
            ob, Dcfg, pw, Dcfg, pb, h, Dcfg, h, Dcfg, Dcfg, Dcfg);

        // n = LN2(h)
        ln_kernel<<<Mrows, 256, 0, stream>>>(h, ln2_w + l * Dcfg, ln2_b + l * Dcfg, nbuf);

        // m = gelu(n @ fw + fb)    [2048,768]x[768,3072]
        gemm_wmma_kernel<0, 1, 0><<<dim3(4 * Dcfg / 256, Mrows / 16), blk128, 0, stream>>>(
            nbuf, Dcfg, fw, 4 * Dcfg, fb, nullptr, 0, mb, 4 * Dcfg, 4 * Dcfg, Dcfg);

        // h = h + m @ f2w + f2b    [2048,3072]x[3072,768]
        gemm_wmma_kernel<0, 0, 1><<<dim3(Dcfg / 256, Mrows / 16), blk128, 0, stream>>>(
            mb, 4 * Dcfg, f2w, Dcfg, f2b, h, Dcfg, h, Dcfg, Dcfg, 4 * Dcfg);
    }

    // Final LN + tied-embedding head: logits = LNf(h) @ wte^T   (BT=1)
    ln_kernel<<<Mrows, 256, 0, stream>>>(h, lnf_w, lnf_b, nbuf);
    gemm_wmma_kernel<1, 0, 0><<<dim3((Vcfg + 255) / 256, Mrows / 16), blk128, 0, stream>>>(
        nbuf, Dcfg, wte, Dcfg, nullptr, nullptr, 0, logits, Vcfg, Vcfg, Dcfg);
}
